// GreedyDecoder_76510547411392
// MI455X (gfx1250) — matrix operand order repaired and checked by reading, not compiled, not run on hardware
//
#include <hip/hip_runtime.h>

// ---------------- problem constants ----------------
#define Tt      512
#define Nn      64
#define Ee      1024
#define Hh      320
#define Jj      512
#define Vv      29
#define BLANKSYM 28
#define MAXSYM  30
#define G4H     1280          // 4*H
#define KC      640           // combined K = [x | h]
#define RESW    15360         // MAX_SYM * T

typedef __bf16 bf16;
typedef __attribute__((ext_vector_type(16))) __bf16 v16bf;
typedef __attribute__((ext_vector_type(8)))  __bf16 v8bf;
typedef __attribute__((ext_vector_type(8)))  float  v8f;

__device__ __forceinline__ float sigm(float x) { return 1.0f / (1.0f + __expf(-x)); }

// ---- WMMA fragment loaders (16x16x32 bf16, wave32; this code's contiguous k order, same for A and B) ----
// A (16x32, 16-bit): lanes 0-15 -> row=lane, K 0..15; lanes 16-31 -> row=lane-16, K 16..31
__device__ __forceinline__ v16bf load_a_frag(const bf16* base, int ld, int m0, int k0, int lane) {
  int row  = m0 + (lane & 15);
  int koff = (lane & 16) ? 16 : 0;
  const bf16* p = base + row * ld + k0 + koff;
  v8bf lo = *(const v8bf*)(p);
  v8bf hi = *(const v8bf*)(p + 8);
  return __builtin_shufflevector(lo, hi, 0,1,2,3,4,5,6,7,8,9,10,11,12,13,14,15);
}
// B (32x16, 16-bit): lanes 0-15 -> col=lane, K 0..15; lanes 16-31 -> col=lane-16, K 16..31
// Weights stored row-major [N][K] (w @ x == x @ w.T), so each lane reads 16 contiguous bf16.
__device__ __forceinline__ v16bf load_b_frag(const bf16* __restrict__ w, int ldk, int n0, int k0, int lane) {
  int col  = n0 + (lane & 15);
  int koff = (lane & 16) ? 16 : 0;
  return *(const v16bf*)(w + (size_t)col * ldk + k0 + koff);
}

__device__ __forceinline__ v8f wmma_bf16(v16bf a, v16bf b, v8f c) {
  return __builtin_amdgcn_wmma_f32_16x16x32_bf16(false, a, false, b, (short)0, c, false, false);
}

// M=64 x N=(ntiles*16) GEMM, K=ksteps*32.  A in LDS (bf16), B in global (bf16, row-major [N][K]).
// Each wave owns N-tiles nt = wave, wave+32, ... with 4 M-tile accumulators -> B loaded once per kstep.
__device__ __forceinline__ void gemm_MxN(const bf16* Alds, int lda,
                                         const bf16* __restrict__ Bgl, int ldk,
                                         int ksteps, int ntiles,
                                         const float* __restrict__ bias,
                                         float* __restrict__ Cout, int ldc,
                                         int wave, int lane) {
  for (int nt = wave; nt < ntiles; nt += 32) {
    v8f z = {0.f,0.f,0.f,0.f,0.f,0.f,0.f,0.f};
    v8f a0 = z, a1 = z, a2 = z, a3 = z;
    for (int kk = 0; kk < ksteps; ++kk) {
      v16bf b = load_b_frag(Bgl, ldk, nt*16, kk*32, lane);
      a0 = wmma_bf16(load_a_frag(Alds, lda,  0, kk*32, lane), b, a0);
      a1 = wmma_bf16(load_a_frag(Alds, lda, 16, kk*32, lane), b, a1);
      a2 = wmma_bf16(load_a_frag(Alds, lda, 32, kk*32, lane), b, a2);
      a3 = wmma_bf16(load_a_frag(Alds, lda, 48, kk*32, lane), b, a3);
    }
    int col = nt*16 + (lane & 15);
    float bv = bias[col];
    int rb = (lane & 16) ? 8 : 0;
#pragma unroll
    for (int r = 0; r < 8; ++r) {       // D layout: VGPR r -> M = r (+8 for lanes>=16)
      Cout[( 0 + rb + r)*ldc + col] = a0[r] + bv;
      Cout[(16 + rb + r)*ldc + col] = a1[r] + bv;
      Cout[(32 + rb + r)*ldc + col] = a2[r] + bv;
      Cout[(48 + rb + r)*ldc + col] = a3[r] + bv;
    }
  }
}

// ---------------- prep: weight conversion / packing + output init ----------------
__global__ void prep_kernel(const float* __restrict__ emb,
                            const float* __restrict__ wi0, const float* __restrict__ wh0,
                            const float* __restrict__ bi0, const float* __restrict__ bh0,
                            const float* __restrict__ wi1, const float* __restrict__ wh1,
                            const float* __restrict__ bi1, const float* __restrict__ bh1,
                            const float* __restrict__ w_f, const float* __restrict__ w_g,
                            const float* __restrict__ b_j, const float* __restrict__ w_o,
                            const float* __restrict__ b_o,
                            bf16* __restrict__ wc0, bf16* __restrict__ wc1,
                            bf16* __restrict__ wf,  bf16* __restrict__ wg,
                            bf16* __restrict__ embbf,
                            float* __restrict__ b0, float* __restrict__ b1,
                            float* __restrict__ bj, float* __restrict__ bo,
                            float* __restrict__ wo, int* __restrict__ out_res) {
  const long stride = (long)gridDim.x * blockDim.x;
  const long t0 = (long)blockIdx.x * blockDim.x + threadIdx.x;
  for (long i = t0; i < (long)G4H*KC; i += stride) {       // combined [4H][x|h] bf16
    int r = (int)(i / KC), k = (int)(i % KC);
    wc0[i] = (bf16)((k < Hh) ? wi0[r*Hh + k] : wh0[r*Hh + k - Hh]);
    wc1[i] = (bf16)((k < Hh) ? wi1[r*Hh + k] : wh1[r*Hh + k - Hh]);
  }
  for (long i = t0; i < (long)Jj*Ee; i += stride) wf[i] = (bf16)w_f[i];
  for (long i = t0; i < (long)Jj*Hh; i += stride) wg[i] = (bf16)w_g[i];
  for (long i = t0; i < (long)(Vv-1)*Hh; i += stride) embbf[i] = (bf16)emb[i];
  for (long i = t0; i < G4H; i += stride) { b0[i] = bi0[i] + bh0[i]; b1[i] = bi1[i] + bh1[i]; }
  for (long i = t0; i < Jj; i += stride) bj[i] = b_j[i];
  for (long i = t0; i < 32; i += stride) bo[i] = (i < Vv) ? b_o[i] : -1e30f;  // pad: never argmax
  for (long i = t0; i < 32L*Jj; i += stride) { int v = (int)(i / Jj); wo[i] = (v < Vv) ? w_o[i] : 0.f; }
  for (long i = t0; i < (long)Nn*RESW; i += stride) out_res[i] = -1;
}

// ---------------- persistent single-WGP decoder ----------------
__global__ void __launch_bounds__(1024)
rnnt_decode(const float* __restrict__ f, const int* __restrict__ f_lens,
            const bf16* __restrict__ wc0, const bf16* __restrict__ wc1,
            const bf16* __restrict__ wf,  const bf16* __restrict__ wg,
            const bf16* __restrict__ embbf,
            const float* __restrict__ b0, const float* __restrict__ b1,
            const float* __restrict__ bj, const float* __restrict__ bo,
            const float* __restrict__ wo,
            float* __restrict__ gates, float* __restrict__ hj,
            bf16* __restrict__ hstate, bf16* __restrict__ hnew,
            float* __restrict__ cstate, float* __restrict__ cnew,
            int* __restrict__ out_res, int* __restrict__ out_cnt)
{
  __shared__ __align__(32) bf16 s_abuf[Nn * KC];   // 80KB: A-tiles / g-buffer / fi-chunk
  __shared__ float s_logits[Nn * 32];
  __shared__ int s_time[Nn], s_symadd[Nn], s_residx[Nn], s_last[Nn],
                 s_flen[Nn], s_emit[Nn], s_tclip[Nn];
  __shared__ int s_cont;

  const int tid  = threadIdx.x;
  const int wave = tid >> 5;
  const int lane = tid & 31;
  bf16* fibuf = s_abuf + Nn * Hh;    // [64][64] bf16 chunk of encoder rows (disjoint from gbuf)
  bf16* gbuf  = s_abuf;              // [64][320] g = h1_new during joint

  for (int i = tid; i < 2*Nn*Hh; i += 1024) { hstate[i] = (bf16)0.0f; cstate[i] = 0.0f; }
  if (tid < Nn) { s_time[tid]=0; s_symadd[tid]=0; s_residx[tid]=-1; s_last[tid]=-1;
                  s_flen[tid]=f_lens[tid]; s_emit[tid]=0; }
  if (tid == 0) s_cont = 1;
  __threadfence();
  __syncthreads();

  for (int iter = 0; iter < 16000; ++iter) {
    if (!s_cont) break;

    // ---- stage A0 = [x | h0_prev] into LDS; clip time indices
    if (tid < Nn) s_tclip[tid] = (s_time[tid] < Tt-1) ? s_time[tid] : (Tt-1);
#pragma unroll
    for (int s = 0; s < 20; ++s) {
      int e = s*1024 + tid;
      int n = e / Hh, k = e - n*Hh;
      int lst = s_last[n];
      s_abuf[n*KC + k]      = (lst < 0) ? (bf16)0.0f : embbf[lst*Hh + k];
      s_abuf[n*KC + Hh + k] = hstate[n*Hh + k];
    }
    __syncthreads();

    // ---- LSTM layer 0: gates = [x|h0] @ wc0^T + b0   (WMMA bf16)
    gemm_MxN(s_abuf, KC, wc0, KC, KC/32, G4H/16, b0, gates, G4H, wave, lane);
    __threadfence(); __syncthreads();

    // ---- elementwise cell 0; stage A1 = [h0_new | h1_prev]
#pragma unroll
    for (int s = 0; s < 20; ++s) {
      int e = s*1024 + tid;
      int n = e / Hh, k = e - n*Hh;
      const float* gr = gates + n*G4H;
      float c2 = sigm(gr[Hh+k]) * cstate[n*Hh+k] + sigm(gr[k]) * tanhf(gr[2*Hh+k]);
      float h  = sigm(gr[3*Hh+k]) * tanhf(c2);
      cnew[n*Hh + k] = c2;
      bf16 hb = (bf16)h;
      hnew[n*Hh + k] = hb;
      s_abuf[n*KC + k]      = hb;
      s_abuf[n*KC + Hh + k] = hstate[Nn*Hh + n*Hh + k];
    }
    __threadfence(); __syncthreads();

    // ---- LSTM layer 1
    gemm_MxN(s_abuf, KC, wc1, KC, KC/32, G4H/16, b1, gates, G4H, wave, lane);
    __threadfence(); __syncthreads();

    // ---- elementwise cell 1: g = h1_new -> gbuf (LDS) + hnew[1]
#pragma unroll
    for (int s = 0; s < 20; ++s) {
      int e = s*1024 + tid;
      int n = e / Hh, k = e - n*Hh;
      const float* gr = gates + n*G4H;
      float c2 = sigm(gr[Hh+k]) * cstate[Nn*Hh + n*Hh+k] + sigm(gr[k]) * tanhf(gr[2*Hh+k]);
      float h  = sigm(gr[3*Hh+k]) * tanhf(c2);
      cnew[Nn*Hh + n*Hh + k] = c2;
      bf16 hb = (bf16)h;
      hnew[Nn*Hh + n*Hh + k] = hb;
      gbuf[n*Hh + k] = hb;
    }
    __threadfence(); __syncthreads();

    // ---- joint: hj = relu(fi @ wf^T + g @ wg^T + bj); wave w owns N-tile w (J/16 == 32)
    {
      v8f z = {0.f,0.f,0.f,0.f,0.f,0.f,0.f,0.f};
      v8f j0 = z, j1 = z, j2 = z, j3 = z;
      for (int ch = 0; ch < Ee/64; ++ch) {        // stream encoder rows in 64-wide bf16 chunks
#pragma unroll
        for (int s2 = 0; s2 < 4; ++s2) {
          int e = s2*1024 + tid;
          int n = e >> 6, kc = e & 63;
          fibuf[n*64 + kc] = (bf16)f[((size_t)s_tclip[n]*Nn + n)*Ee + ch*64 + kc];
        }
        __syncthreads();
#pragma unroll
        for (int kk = 0; kk < 2; ++kk) {
          v16bf b = load_b_frag(wf, Ee, wave*16, ch*64 + kk*32, lane);
          j0 = wmma_bf16(load_a_frag(fibuf, 64,  0, kk*32, lane), b, j0);
          j1 = wmma_bf16(load_a_frag(fibuf, 64, 16, kk*32, lane), b, j1);
          j2 = wmma_bf16(load_a_frag(fibuf, 64, 32, kk*32, lane), b, j2);
          j3 = wmma_bf16(load_a_frag(fibuf, 64, 48, kk*32, lane), b, j3);
        }
        __syncthreads();
      }
#pragma unroll
      for (int kk = 0; kk < Hh/32; ++kk) {        // g contribution from LDS
        v16bf b = load_b_frag(wg, Hh, wave*16, kk*32, lane);
        j0 = wmma_bf16(load_a_frag(gbuf, Hh,  0, kk*32, lane), b, j0);
        j1 = wmma_bf16(load_a_frag(gbuf, Hh, 16, kk*32, lane), b, j1);
        j2 = wmma_bf16(load_a_frag(gbuf, Hh, 32, kk*32, lane), b, j2);
        j3 = wmma_bf16(load_a_frag(gbuf, Hh, 48, kk*32, lane), b, j3);
      }
      int col = wave*16 + (lane & 15);
      float bv = bj[col];
      int rb = (lane & 16) ? 8 : 0;
#pragma unroll
      for (int r = 0; r < 8; ++r) {
        float v0 = j0[r] + bv; hj[( 0+rb+r)*Jj + col] = v0 > 0.f ? v0 : 0.f;
        float v1 = j1[r] + bv; hj[(16+rb+r)*Jj + col] = v1 > 0.f ? v1 : 0.f;
        float v2 = j2[r] + bv; hj[(32+rb+r)*Jj + col] = v2 > 0.f ? v2 : 0.f;
        float v3 = j3[r] + bv; hj[(48+rb+r)*Jj + col] = v3 > 0.f ? v3 : 0.f;
      }
    }
    __threadfence(); __syncthreads();

    // ---- logits: tiny f32 GEMV (64x29, K=512) kept full precision for argmax fidelity
    for (int e = tid; e < Nn*Vv; e += 1024) {
      int n = e / Vv, v = e - n*Vv;
      const float* hr = hj + n*Jj;
      const float* wr = wo + v*Jj;
      float sacc = bo[v];
      for (int j = 0; j < Jj; ++j) sacc += hr[j] * wr[j];
      s_logits[n*32 + v] = sacc;
    }
    __syncthreads();

    // ---- greedy control (one thread per batch row)
    if (tid < Nn) {
      int n = tid;
      float best = s_logits[n*32]; int bsym = 0;
      for (int v = 1; v < Vv; ++v) { float lv = s_logits[n*32 + v]; if (lv > best) { best = lv; bsym = v; } }
      bool active = s_time[n] < s_flen[n];
      bool emit = active && (bsym != BLANKSYM) && (s_symadd[n] < MAXSYM);
      bool adv  = active && !emit;
      if (emit) {
        int slot = s_residx[n] + 1;
        out_res[n*RESW + slot] = bsym;
        s_residx[n] = slot;
        s_symadd[n] += 1;
        s_last[n] = bsym;
      } else if (adv) {
        s_symadd[n] = 0;
        s_time[n] += 1;
      }
      s_emit[n] = emit ? 1 : 0;
    }
    if (tid == 0) s_cont = 0;
    __syncthreads();
    if (tid < Nn && s_time[tid] < s_flen[tid]) s_cont = 1;

    // ---- commit h/c state only where a symbol was emitted
#pragma unroll
    for (int s = 0; s < 40; ++s) {
      int e = s*1024 + tid;                 // flattened [2][N][H]
      int n = (e / Hh) % Nn;
      if (s_emit[n]) { hstate[e] = hnew[e]; cstate[e] = cnew[e]; }
    }
    __threadfence(); __syncthreads();
  }

  if (tid < Nn) out_cnt[tid] = s_residx[tid] + 1;
}

// ---------------- host launch ----------------
extern "C" void kernel_launch(void* const* d_in, const int* in_sizes, int n_in,
                              void* d_out, int out_size, void* d_ws, size_t ws_size,
                              hipStream_t stream) {
  const float* f     = (const float*)d_in[0];
  const int*   flens = (const int*)  d_in[1];
  const float* emb   = (const float*)d_in[2];
  const float* wi0   = (const float*)d_in[3];
  const float* wh0   = (const float*)d_in[4];
  const float* bi0   = (const float*)d_in[5];
  const float* bh0   = (const float*)d_in[6];
  const float* wi1   = (const float*)d_in[7];
  const float* wh1   = (const float*)d_in[8];
  const float* bi1   = (const float*)d_in[9];
  const float* bh1   = (const float*)d_in[10];
  const float* w_f   = (const float*)d_in[11];
  const float* w_g   = (const float*)d_in[12];
  const float* b_j   = (const float*)d_in[13];
  const float* w_o   = (const float*)d_in[14];
  const float* b_o   = (const float*)d_in[15];

  char* ws = (char*)d_ws;
  size_t off = 0;
  auto take = [&](size_t bytes) { size_t o = off; off += (bytes + 255) & ~(size_t)255; return o; };
  bf16*  wc0    = (bf16*)(ws + take((size_t)G4H*KC*2));
  bf16*  wc1    = (bf16*)(ws + take((size_t)G4H*KC*2));
  bf16*  wfb    = (bf16*)(ws + take((size_t)Jj*Ee*2));
  bf16*  wgb    = (bf16*)(ws + take((size_t)Jj*Hh*2));
  bf16*  embbf  = (bf16*)(ws + take((size_t)(Vv-1)*Hh*2));
  float* b0     = (float*)(ws + take((size_t)G4H*4));
  float* b1     = (float*)(ws + take((size_t)G4H*4));
  float* bjb    = (float*)(ws + take((size_t)Jj*4));
  float* bob    = (float*)(ws + take(32*4));
  float* wob    = (float*)(ws + take((size_t)32*Jj*4));
  bf16*  hstate = (bf16*)(ws + take((size_t)2*Nn*Hh*2));
  bf16*  hnew   = (bf16*)(ws + take((size_t)2*Nn*Hh*2));
  float* cstate = (float*)(ws + take((size_t)2*Nn*Hh*4));
  float* cnew   = (float*)(ws + take((size_t)2*Nn*Hh*4));
  float* gates  = (float*)(ws + take((size_t)Nn*G4H*4));
  float* hjb    = (float*)(ws + take((size_t)Nn*Jj*4));
  if (ws_size < off) return;

  int* out_res = (int*)d_out;
  int* out_cnt = out_res + (size_t)Nn*RESW;

  prep_kernel<<<1024, 256, 0, stream>>>(emb, wi0, wh0, bi0, bh0, wi1, wh1, bi1, bh1,
                                        w_f, w_g, b_j, w_o, b_o,
                                        wc0, wc1, wfb, wgb, embbf, b0, b1, bjb, bob, wob,
                                        out_res);
  rnnt_decode<<<1, 1024, 0, stream>>>(f, flens, wc0, wc1, wfb, wgb, embbf,
                                      b0, b1, bjb, bob, wob,
                                      gates, hjb, hstate, hnew, cstate, cnew,
                                      out_res, out_cnt);
}
